// Intra_Modal_Modeling_19215683682433
// MI455X (gfx1250) — compile-verified, matrix operand order checked by reading, not run on hardware
//
#include <hip/hip_runtime.h>
#include <hip/hip_bf16.h>
#include <math.h>

// ---------------- problem dimensions ----------------
#define BB    16
#define NNN   1024
#define EEE   256
#define HHH   4
#define DHH   64
#define HIDD  512
#define LLL   3
#define MMM   (BB * NNN)          /* 16384 token rows */
#define NEGB  (-1e9f)

typedef __attribute__((ext_vector_type(16))) __bf16 v16bf;
typedef __attribute__((ext_vector_type(8)))  float  v8f;
typedef unsigned short ushort_t;

union FragA { v16bf v; unsigned short s[16]; unsigned u[8]; };
union FragC { v8f   v; float f[8]; };

// hardware f32 -> bf16 convert (single v_cvt instead of integer RNE sequence)
__device__ __forceinline__ unsigned short f2bf(float x) {
  __bf16 h = (__bf16)x;
  return __builtin_bit_cast(unsigned short, h);
}

// K-index pattern for 16-bit A/B fragments (ISA 7.12.2): VGPR t<4 -> K=2t,
// t>=4 -> K=16+2(t-4); +8 for upper half-wave.
__device__ __forceinline__ int kpat(int t, int half) {
  return ((t < 4) ? (2 * t) : (16 + 2 * (t - 4))) + 8 * half;
}

// Load one fragment from a bf16 buffer (K-pairs contiguous): pure dword loads.
__device__ __forceinline__ void load_frag16(FragA& fr, const ushort_t* __restrict__ p,
                                            int half) {
#pragma unroll
  for (int t = 0; t < 8; ++t)
    fr.u[t] = *(const unsigned*)(p + kpat(t, half));
}

// ============================================================
// Generic GEMM on bf16 operands:
//   D[M x Nout] = act(A[M x K] * W[Nout x K]^T + bias)
// A, W in bf16; D (f32) and D16 (bf16) outputs each optional.
// block = 8 waves -> 128(M) x 64(N); wave = 32x32 via 2x2 WMMA.
// ============================================================
__global__ __launch_bounds__(256)
void gemm_kernel(const ushort_t* __restrict__ A, const ushort_t* __restrict__ W,
                 const float* __restrict__ bias, float* __restrict__ D,
                 ushort_t* __restrict__ D16, int Nout, int K, int act)
{
  const int lane = threadIdx.x & 31;
  const int wave = threadIdx.x >> 5;
  const int half = lane >> 4;
  const int l16  = lane & 15;
  const int mbase = blockIdx.x * 128 + (wave & 3) * 32;
  const int nbase = blockIdx.y * 64 + (wave >> 2) * 32;

  FragC c[2][2];
#pragma unroll
  for (int mt = 0; mt < 2; ++mt)
#pragma unroll
    for (int nt = 0; nt < 2; ++nt)
#pragma unroll
      for (int r = 0; r < 8; ++r) c[mt][nt].f[r] = 0.f;

  for (int k0 = 0; k0 < K; k0 += 32) {
    FragA a[2], b[2];
#pragma unroll
    for (int mt = 0; mt < 2; ++mt)
      load_frag16(a[mt], A + (size_t)(mbase + mt * 16 + l16) * K + k0, half);
#pragma unroll
    for (int nt = 0; nt < 2; ++nt)
      load_frag16(b[nt], W + (size_t)(nbase + nt * 16 + l16) * K + k0, half);
#pragma unroll
    for (int mt = 0; mt < 2; ++mt)
#pragma unroll
      for (int nt = 0; nt < 2; ++nt)
        c[mt][nt].v = __builtin_amdgcn_wmma_f32_16x16x32_bf16(
            false, a[mt].v, false, b[nt].v, (short)0, c[mt][nt].v, false, false);
  }

#pragma unroll
  for (int mt = 0; mt < 2; ++mt)
#pragma unroll
    for (int nt = 0; nt < 2; ++nt)
#pragma unroll
      for (int r = 0; r < 8; ++r) {
        int row = mbase + mt * 16 + r + 8 * half;
        int col = nbase + nt * 16 + l16;
        float v = c[mt][nt].f[r] + bias[col];
        if (act == 1) v = fmaxf(v, 0.f);
        size_t idx = (size_t)row * Nout + col;
        if (D)   D[idx] = v;
        if (D16) D16[idx] = f2bf(v);
      }
}

// ============================================================
// Flash attention over bf16 Q/K/V ([B,N,E] with head h at cols [h*64,h*64+64)).
// One wave = 16-query tile of one (b,h); keys streamed 32 at a time.
// mode 0: key-padding bias from `pad`; mode 1/2: allowed-bitmask bias.
// Output: bf16 (feeds out_proj GEMM only).
// ============================================================
__global__ __launch_bounds__(128)
void attn_kernel(const ushort_t* __restrict__ Q, const ushort_t* __restrict__ Km,
                 const ushort_t* __restrict__ V, const unsigned char* __restrict__ pad,
                 const unsigned* __restrict__ allow, int mode,
                 ushort_t* __restrict__ O16)
{
  __shared__ unsigned short pbuf[4][16][32];   // per-wave P relayout tile
  const int wv   = threadIdx.x >> 5;
  const int lane = threadIdx.x & 31;
  const int half = lane >> 4;
  const int l16  = lane & 15;
  const int gw   = blockIdx.x * 4 + wv;        // 0 .. B*H*(N/16)-1
  const int qt   = gw & 63;
  const int h    = (gw >> 6) & 3;
  const int b    = gw >> 8;
  const int q0   = qt * 16;
  const size_t base = (size_t)b * NNN * EEE + (size_t)h * DHH;

  FragA qa[2];
#pragma unroll
  for (int ks = 0; ks < 2; ++ks)
    load_frag16(qa[ks], Q + base + (size_t)(q0 + l16) * EEE + ks * 32, half);

  FragC o[4];
#pragma unroll
  for (int j = 0; j < 4; ++j)
#pragma unroll
    for (int r = 0; r < 8; ++r) o[j].f[r] = 0.f;
  float mrow[8], lrow[8];
#pragma unroll
  for (int r = 0; r < 8; ++r) { mrow[r] = -1e30f; lrow[r] = 0.f; }

  for (int jj = 0; jj < NNN; jj += 32) {
    // ---- S = Q * K^T (two 16-key tiles, DH contracted in 2 steps) ----
    FragC s0, s1;
#pragma unroll
    for (int r = 0; r < 8; ++r) { s0.f[r] = 0.f; s1.f[r] = 0.f; }
#pragma unroll
    for (int ks = 0; ks < 2; ++ks) {
      FragA kf0, kf1;
      load_frag16(kf0, Km + base + (size_t)(jj + l16) * EEE + ks * 32, half);
      load_frag16(kf1, Km + base + (size_t)(jj + 16 + l16) * EEE + ks * 32, half);
      s0.v = __builtin_amdgcn_wmma_f32_16x16x32_bf16(false, qa[ks].v, false, kf0.v,
                                                     (short)0, s0.v, false, false);
      s1.v = __builtin_amdgcn_wmma_f32_16x16x32_bf16(false, qa[ks].v, false, kf1.v,
                                                     (short)0, s1.v, false, false);
    }

    float b0 = 0.f, b1 = 0.f;
    if (mode == 0) {
      b0 = pad[b * NNN + jj + l16] ? NEGB : 0.f;
      b1 = pad[b * NNN + jj + 16 + l16] ? NEGB : 0.f;
    }

    // ---- online softmax over the 32 new scores of each row ----
#pragma unroll
    for (int r = 0; r < 8; ++r) {
      float bb0 = b0, bb1 = b1;
      if (mode != 0) {
        int qrow = q0 + r + 8 * half;
        unsigned wrd = allow[((size_t)b * NNN + qrow) * 32 + (jj >> 5)];
        bb0 = ((wrd >> l16) & 1u)        ? 0.f : NEGB;
        bb1 = ((wrd >> (16 + l16)) & 1u) ? 0.f : NEGB;
      }
      float v0 = s0.f[r] * 0.125f + bb0;   // 1/sqrt(64)
      float v1 = s1.f[r] * 0.125f + bb1;
      float mx = fmaxf(v0, v1);
#pragma unroll
      for (int off = 1; off < 16; off <<= 1) mx = fmaxf(mx, __shfl_xor(mx, off, 16));
      float mnew  = fmaxf(mrow[r], mx);
      float alpha = __expf(mrow[r] - mnew);
      float p0 = __expf(v0 - mnew), p1 = __expf(v1 - mnew);
      float ps = p0 + p1;
#pragma unroll
      for (int off = 1; off < 16; off <<= 1) ps += __shfl_xor(ps, off, 16);
      lrow[r] = lrow[r] * alpha + ps;
      mrow[r] = mnew;
#pragma unroll
      for (int j = 0; j < 4; ++j) o[j].f[r] *= alpha;
      int row = r + 8 * half;
      pbuf[wv][row][l16]      = f2bf(p0);
      pbuf[wv][row][16 + l16] = f2bf(p1);
    }
    __syncthreads();

    // ---- P relayout (LDS) -> A fragment ----
    FragA pa;
#pragma unroll
    for (int t = 0; t < 8; ++t) {
      int kb = kpat(t, half);
      pa.u[t] = *(const unsigned*)&pbuf[wv][l16][kb];
    }
    __syncthreads();

    // ---- O += P * V (V already bf16: raw b16 loads, no conversion) ----
#pragma unroll
    for (int nt = 0; nt < 4; ++nt) {
      FragA vb;
#pragma unroll
      for (int t = 0; t < 8; ++t) {
        int kb = kpat(t, half);
        const ushort_t* vp = V + base + (size_t)(jj + kb) * EEE + nt * 16 + l16;
        vb.s[2 * t]     = vp[0];
        vb.s[2 * t + 1] = vp[EEE];
      }
      o[nt].v = __builtin_amdgcn_wmma_f32_16x16x32_bf16(false, pa.v, false, vb.v,
                                                        (short)0, o[nt].v, false, false);
    }
  }

#pragma unroll
  for (int nt = 0; nt < 4; ++nt)
#pragma unroll
    for (int r = 0; r < 8; ++r) {
      int row = q0 + r + 8 * half;
      O16[base + (size_t)row * EEE + nt * 16 + l16] = f2bf(o[nt].f[r] / lrow[r]);
    }
}

// ============================================================
// Per-batch cosine similarity: S[b] = clip((nv[b] . nv[b]^T + 1)/2, 0, 1)
// nv is [B, N, 64] bf16 (L2-normalized rows). grid=(N/128, N/64, B).
// ============================================================
__global__ __launch_bounds__(256)
void sim_kernel(const ushort_t* __restrict__ NV, float* __restrict__ S)
{
  const int lane = threadIdx.x & 31;
  const int wave = threadIdx.x >> 5;
  const int half = lane >> 4;
  const int l16  = lane & 15;
  const int bidx = blockIdx.z;
  const ushort_t* A = NV + (size_t)bidx * NNN * 64;
  const int mbase = blockIdx.x * 128 + (wave & 3) * 32;
  const int nbase = blockIdx.y * 64 + (wave >> 2) * 32;

  FragC c[2][2];
#pragma unroll
  for (int mt = 0; mt < 2; ++mt)
#pragma unroll
    for (int nt = 0; nt < 2; ++nt)
#pragma unroll
      for (int r = 0; r < 8; ++r) c[mt][nt].f[r] = 0.f;

#pragma unroll
  for (int k0 = 0; k0 < 64; k0 += 32) {
    FragA a[2], b[2];
#pragma unroll
    for (int mt = 0; mt < 2; ++mt)
      load_frag16(a[mt], A + (size_t)(mbase + mt * 16 + l16) * 64 + k0, half);
#pragma unroll
    for (int nt = 0; nt < 2; ++nt)
      load_frag16(b[nt], A + (size_t)(nbase + nt * 16 + l16) * 64 + k0, half);
#pragma unroll
    for (int mt = 0; mt < 2; ++mt)
#pragma unroll
      for (int nt = 0; nt < 2; ++nt)
        c[mt][nt].v = __builtin_amdgcn_wmma_f32_16x16x32_bf16(
            false, a[mt].v, false, b[nt].v, (short)0, c[mt][nt].v, false, false);
  }

#pragma unroll
  for (int mt = 0; mt < 2; ++mt)
#pragma unroll
    for (int nt = 0; nt < 2; ++nt)
#pragma unroll
      for (int r = 0; r < 8; ++r) {
        int row = mbase + mt * 16 + r + 8 * half;
        int col = nbase + nt * 16 + l16;
        float v = (c[mt][nt].f[r] + 1.f) * 0.5f;
        v = fminf(fmaxf(v, 0.f), 1.f);
        S[(size_t)bidx * NNN * NNN + (size_t)row * NNN + col] = v;
      }
}

// ============================================================
// LayerNorm (optional residual-add before, gelu after, dual f32/bf16 out).
// One wave per row.
// ============================================================
__global__ __launch_bounds__(256)
void ln_kernel(const float* __restrict__ X, const float* __restrict__ Res,
               const float* __restrict__ g, const float* __restrict__ be,
               float* __restrict__ Y, ushort_t* __restrict__ Y16, int dim, int gelu)
{
  const int gw   = blockIdx.x * 8 + (threadIdx.x >> 5);
  const int lane = threadIdx.x & 31;
  const float* xr = X + (size_t)gw * dim;
  const float* rr = Res ? Res + (size_t)gw * dim : nullptr;
  const int cnt = dim >> 5;               // 4 / 8 / 16
  float v[16];
  float sum = 0.f;
  for (int j = 0; j < cnt; ++j) {
    int c = lane + 32 * j;
    float t = xr[c];
    if (rr) t += rr[c];
    v[j] = t; sum += t;
  }
#pragma unroll
  for (int off = 1; off < 32; off <<= 1) sum += __shfl_xor(sum, off, 32);
  float mean = sum / (float)dim;
  float var = 0.f;
  for (int j = 0; j < cnt; ++j) { float d = v[j] - mean; var += d * d; }
#pragma unroll
  for (int off = 1; off < 32; off <<= 1) var += __shfl_xor(var, off, 32);
  float inv = rsqrtf(var / (float)dim + 1e-5f);
  for (int j = 0; j < cnt; ++j) {
    int c = lane + 32 * j;
    float y = (v[j] - mean) * inv * g[c] + be[c];
    if (gelu) y = 0.5f * y * (1.f + erff(y * 0.70710678118f));
    if (Y)   Y[(size_t)gw * dim + c] = y;
    if (Y16) Y16[(size_t)gw * dim + c] = f2bf(y);
  }
}

// elementwise c = a + b, dual f32 / bf16 outputs (each optional)
__global__ void add_kernel(const float* __restrict__ a, const float* __restrict__ b,
                           float* __restrict__ c, ushort_t* __restrict__ c16, int n)
{
  int i = blockIdx.x * blockDim.x + threadIdx.x;
  if (i < n) {
    float v = a[i] + b[i];
    if (c)   c[i] = v;
    if (c16) c16[i] = f2bf(v);
  }
}

// f32 -> bf16 copy
__global__ void cvt_kernel(const float* __restrict__ a, ushort_t* __restrict__ o, int n)
{
  int i = blockIdx.x * blockDim.x + threadIdx.x;
  if (i < n) o[i] = f2bf(a[i]);
}

// L2-normalize rows of [rows, 64] -> bf16. One wave per row.
__global__ __launch_bounds__(256)
void l2norm_kernel(const float* __restrict__ X, ushort_t* __restrict__ Y16)
{
  const int gw   = blockIdx.x * 8 + (threadIdx.x >> 5);
  const int lane = threadIdx.x & 31;
  const float* xr = X + (size_t)gw * 64;
  float v0 = xr[lane], v1 = xr[lane + 32];
  float ss = v0 * v0 + v1 * v1;
#pragma unroll
  for (int off = 1; off < 32; off <<= 1) ss += __shfl_xor(ss, off, 32);
  float inv = rsqrtf(ss);
  Y16[(size_t)gw * 64 + lane]      = f2bf(v0 * inv);
  Y16[(size_t)gw * 64 + lane + 32] = f2bf(v1 * inv);
}

__global__ void mask_any_kernel(const unsigned char* __restrict__ mask, int n,
                                int* __restrict__ flag)
{
  __shared__ int s;
  if (threadIdx.x == 0) s = 0;
  __syncthreads();
  int acc = 0;
  for (int i = threadIdx.x; i < n; i += blockDim.x) acc |= mask[i];
  if (acc) atomicOr(&s, 1);
  __syncthreads();
  if (threadIdx.x == 0) *flag = s;
}

// ============================================================
// Top-32 neighbor selection per row -> "allowed" bitmasks.
// One wave per (b, q) row; 32 values per lane in registers.
// ============================================================
__global__ __launch_bounds__(256)
void topk_kernel(const float* __restrict__ sim, const unsigned char* __restrict__ mm,
                 const int* __restrict__ hasflag,
                 unsigned* __restrict__ allowS, unsigned* __restrict__ allowU)
{
  const int gw   = blockIdx.x * 8 + (threadIdx.x >> 5);
  const int lane = threadIdx.x & 31;
  const int q = gw & (NNN - 1);
  const bool has = hasflag[0] != 0;
  const float* srow = sim + (size_t)gw * NNN;
  const unsigned char* mrow = mm + (size_t)gw * NNN;

  float sv[32], uv[32];
#pragma unroll
  for (int j = 0; j < 32; ++j) {
    int idx = lane + 32 * j;
    float v = srow[idx];
    bool inval = has && (mrow[idx] == 0);
    uv[j] = inval ? 2.0f : v;
    sv[j] = (inval ? -1.0f : v) - ((idx == q) ? 1.0f : 0.0f);
  }

  unsigned chosenU = 0, chosenS = 0;
  for (int it = 0; it < 32; ++it) {                 // 32 x argmin (least similar)
    float best = 3e38f; int bidx = 1 << 30;
#pragma unroll
    for (int j = 0; j < 32; ++j)
      if (!((chosenU >> j) & 1u) && uv[j] < best) { best = uv[j]; bidx = lane + 32 * j; }
#pragma unroll
    for (int off = 1; off < 32; off <<= 1) {
      float ov = __shfl_xor(best, off, 32);
      int   oi = __shfl_xor(bidx, off, 32);
      if (ov < best || (ov == best && oi < bidx)) { best = ov; bidx = oi; }
    }
    if (bidx < (1 << 30) && (bidx & 31) == lane) chosenU |= 1u << (bidx >> 5);
  }
  for (int it = 0; it < 32; ++it) {                 // 32 x argmax (most similar)
    float best = -3e38f; int bidx = 1 << 30;
#pragma unroll
    for (int j = 0; j < 32; ++j)
      if (!((chosenS >> j) & 1u) && sv[j] > best) { best = sv[j]; bidx = lane + 32 * j; }
#pragma unroll
    for (int off = 1; off < 32; off <<= 1) {
      float ov = __shfl_xor(best, off, 32);
      int   oi = __shfl_xor(bidx, off, 32);
      if (ov > best || (ov == best && oi < bidx)) { best = ov; bidx = oi; }
    }
    if (bidx < (1 << 30) && (bidx & 31) == lane) chosenS |= 1u << (bidx >> 5);
  }

  for (int j = 0; j < 32; ++j) {   // (lane,bit) -> word transpose via wave32 ballot
    unsigned long long bu = __ballot((chosenU >> j) & 1);
    unsigned long long bs = __ballot((chosenS >> j) & 1);
    if (lane == j) {
      allowU[(size_t)gw * 32 + j] = (unsigned)bu;
      allowS[(size_t)gw * 32 + j] = (unsigned)bs;
    }
  }
}

// ============================================================
// Host orchestration
// ============================================================
extern "C" void kernel_launch(void* const* d_in, const int* in_sizes, int n_in,
                              void* d_out, int out_size, void* d_ws, size_t ws_size,
                              hipStream_t stream)
{
  (void)in_sizes; (void)n_in; (void)out_size; (void)ws_size;

  int i = 0;
  const float*         feats = (const float*)d_in[i++];
  const unsigned char* mask  = (const unsigned char*)d_in[i++];
  const unsigned char* mmask = (const unsigned char*)d_in[i++];
  const float*         pos   = (const float*)d_in[i++];
  const float* si_in_w  = (const float*)d_in[i++];
  const float* si_in_b  = (const float*)d_in[i++];
  const float* si_out_w = (const float*)d_in[i++];
  const float* si_out_b = (const float*)d_in[i++];
  const float* si_ln1_g = (const float*)d_in[i++];
  const float* si_ln1_b = (const float*)d_in[i++];
  const float* si_fw1   = (const float*)d_in[i++];
  const float* si_fb1   = (const float*)d_in[i++];
  const float* si_fw2   = (const float*)d_in[i++];
  const float* si_fb2   = (const float*)d_in[i++];
  const float* si_ln2_g = (const float*)d_in[i++];
  const float* si_ln2_b = (const float*)d_in[i++];
  const float* ce_w1 = (const float*)d_in[i++];
  const float* ce_b1 = (const float*)d_in[i++];
  const float* ce_g1 = (const float*)d_in[i++];
  const float* ce_be1= (const float*)d_in[i++];
  const float* ce_w2 = (const float*)d_in[i++];
  const float* ce_b2 = (const float*)d_in[i++];
  const float* ce_g2 = (const float*)d_in[i++];
  const float* ce_be2= (const float*)d_in[i++];
  const float* ce_w3 = (const float*)d_in[i++];
  const float* ce_b3 = (const float*)d_in[i++];
  const float* a1_in_w  = (const float*)d_in[i++];
  const float* a1_in_b  = (const float*)d_in[i++];
  const float* a1_out_w = (const float*)d_in[i++];
  const float* a1_out_b = (const float*)d_in[i++];
  const float* m1_w1 = (const float*)d_in[i++];
  const float* m1_b1 = (const float*)d_in[i++];
  const float* m1_g1 = (const float*)d_in[i++];
  const float* m1_be1= (const float*)d_in[i++];
  const float* m1_w2 = (const float*)d_in[i++];
  const float* m1_b2 = (const float*)d_in[i++];
  const float* m1_g2 = (const float*)d_in[i++];
  const float* m1_be2= (const float*)d_in[i++];
  const float* m1_w3 = (const float*)d_in[i++];
  const float* m1_b3 = (const float*)d_in[i++];
  const float* a2_in_w  = (const float*)d_in[i++];
  const float* a2_in_b  = (const float*)d_in[i++];
  const float* a2_out_w = (const float*)d_in[i++];
  const float* a2_out_b = (const float*)d_in[i++];
  const float* m2_w1 = (const float*)d_in[i++];
  const float* m2_b1 = (const float*)d_in[i++];
  const float* m2_g1 = (const float*)d_in[i++];
  const float* m2_be1= (const float*)d_in[i++];
  const float* m2_w2 = (const float*)d_in[i++];
  const float* m2_b2 = (const float*)d_in[i++];
  const float* m2_g2 = (const float*)d_in[i++];
  const float* m2_be2= (const float*)d_in[i++];
  const float* m2_w3 = (const float*)d_in[i++];
  const float* m2_b3 = (const float*)d_in[i++];

  // ---- output regions: (feats, sim, consist) concatenated flat ----
  float* outFeats = (float*)d_out;
  float* outSim   = outFeats + (size_t)MMM * EEE;
  float* outCons  = outSim + (size_t)BB * NNN * NNN;

  // ---- workspace (byte allocator, 256B aligned slices) ----
  char* WS = (char*)d_ws;
  size_t off = 0;
  auto takeB = [&](size_t bytes) {
    char* p = WS + off;
    off += (bytes + 255) & ~(size_t)255;
    return (void*)p;
  };
  auto takeF = [&](size_t n) { return (float*)takeB(n * 4); };
  auto takeH = [&](size_t n) { return (ushort_t*)takeB(n * 2); };

  const size_t nE = (size_t)MMM * EEE;        // 16384 x 256
  float* xbuf  = takeF(nE);
  float* aobuf = takeF(nE);
  float* fbuf  = takeF(nE);
  float* t256  = takeF(nE);
  float* hbuf  = takeF((size_t)MMM * HIDD);
  float* hbuf2 = takeF((size_t)MMM * HIDD);
  ushort_t* x16  = takeH(nE);
  ushort_t* qk16 = takeH(nE);
  ushort_t* q16  = takeH(nE);
  ushort_t* k16  = takeH(nE);
  ushort_t* v16  = takeH(nE);
  ushort_t* o16  = takeH(nE);
  ushort_t* ao16 = takeH(nE);
  ushort_t* h16  = takeH((size_t)MMM * HIDD);
  ushort_t* h216 = takeH((size_t)MMM * HIDD);
  ushort_t* t12816 = takeH((size_t)MMM * 128);
  ushort_t* nv16   = takeH((size_t)MMM * 64);
  unsigned* allowS = (unsigned*)takeB((size_t)MMM * 32 * 4);
  unsigned* allowU = (unsigned*)takeB((size_t)MMM * 32 * 4);
  int* hasflag = (int*)takeB(64);
  float* t128 = fbuf;          // fbuf is free during the consist stage
  ushort_t* t25616 = qk16;     // qk16 is free after the encoder layers

  // ---- bf16 weight pool (converted once per launch) ----
  auto cvt = [&](const float* src, size_t n) {
    ushort_t* dst = takeH(n);
    cvt_kernel<<<(int)((n + 255) / 256), 256, 0, stream>>>(src, dst, (int)n);
    return dst;
  };
  ushort_t* w_si_in  = cvt(si_in_w,  (size_t)LLL * 3 * EEE * EEE);
  ushort_t* w_si_out = cvt(si_out_w, (size_t)LLL * EEE * EEE);
  ushort_t* w_si_f1  = cvt(si_fw1,   (size_t)LLL * HIDD * EEE);
  ushort_t* w_si_f2  = cvt(si_fw2,   (size_t)LLL * EEE * HIDD);
  ushort_t* w_ce1 = cvt(ce_w1, 256 * EEE);
  ushort_t* w_ce2 = cvt(ce_w2, 128 * 256);
  ushort_t* w_ce3 = cvt(ce_w3, 64 * 128);
  ushort_t* w_a1_in  = cvt(a1_in_w, 3 * EEE * EEE);
  ushort_t* w_a1_out = cvt(a1_out_w, EEE * EEE);
  ushort_t* w_m1_1 = cvt(m1_w1, 512 * EEE);
  ushort_t* w_m1_2 = cvt(m1_w2, 512 * 512);
  ushort_t* w_m1_3 = cvt(m1_w3, EEE * 512);
  ushort_t* w_a2_in  = cvt(a2_in_w, 3 * EEE * EEE);
  ushort_t* w_a2_out = cvt(a2_out_w, EEE * EEE);
  ushort_t* w_m2_1 = cvt(m2_w1, 512 * EEE);
  ushort_t* w_m2_2 = cvt(m2_w2, 512 * 512);
  ushort_t* w_m2_3 = cvt(m2_w3, EEE * 512);

  const int nElem = MMM * EEE;
  auto gemm = [&](const ushort_t* A, const ushort_t* W, const float* bias,
                  float* D, ushort_t* D16, int Nout, int K, int act) {
    dim3 g(MMM / 128, Nout / 64, 1);
    gemm_kernel<<<g, 256, 0, stream>>>(A, W, bias, D, D16, Nout, K, act);
  };

  hipMemcpyAsync(xbuf, feats, (size_t)nElem * sizeof(float),
                 hipMemcpyDeviceToDevice, stream);
  cvt_kernel<<<nElem / 256, 256, 0, stream>>>(feats, x16, nElem);

  // ---- Self_Interaction: 3 post-LN transformer layers ----
  for (int l = 0; l < LLL; ++l) {
    const ushort_t* inw = w_si_in + (size_t)l * 3 * EEE * EEE;
    const float*    inb = si_in_b + (size_t)l * 3 * EEE;
    add_kernel<<<nElem / 256, 256, 0, stream>>>(xbuf, pos, nullptr, qk16, nElem);
    gemm(qk16, inw,                 inb,           nullptr, q16, EEE, EEE, 0);
    gemm(qk16, inw + EEE * EEE,     inb + EEE,     nullptr, k16, EEE, EEE, 0);
    gemm(x16,  inw + 2 * EEE * EEE, inb + 2 * EEE, nullptr, v16, EEE, EEE, 0);
    attn_kernel<<<1024, 128, 0, stream>>>(q16, k16, v16, mask, nullptr, 0, o16);
    gemm(o16, w_si_out + (size_t)l * EEE * EEE, si_out_b + l * EEE,
         aobuf, nullptr, EEE, EEE, 0);
    ln_kernel<<<MMM / 8, 256, 0, stream>>>(xbuf, aobuf, si_ln1_g + l * EEE,
                                           si_ln1_b + l * EEE, xbuf, x16, EEE, 0);
    gemm(x16, w_si_f1 + (size_t)l * HIDD * EEE, si_fb1 + l * HIDD,
         nullptr, h16, HIDD, EEE, 1);
    gemm(h16, w_si_f2 + (size_t)l * EEE * HIDD, si_fb2 + l * EEE,
         fbuf, nullptr, EEE, HIDD, 0);
    ln_kernel<<<MMM / 8, 256, 0, stream>>>(xbuf, fbuf, si_ln2_g + l * EEE,
                                           si_ln2_b + l * EEE, xbuf, x16, EEE, 0);
  }

  // ---- consist encoder ----
  gemm(x16, w_ce1, ce_b1, t256, nullptr, 256, EEE, 0);
  ln_kernel<<<MMM / 8, 256, 0, stream>>>(t256, nullptr, ce_g1, ce_be1,
                                         nullptr, t25616, 256, 1);
  gemm(t25616, w_ce2, ce_b2, t128, nullptr, 128, 256, 0);
  ln_kernel<<<MMM / 8, 256, 0, stream>>>(t128, nullptr, ce_g2, ce_be2,
                                         nullptr, t12816, 128, 1);
  gemm(t12816, w_ce3, ce_b3, outCons, nullptr, 64, 128, 0);

  // ---- cosine similarity + top-k masks ----
  l2norm_kernel<<<MMM / 8, 256, 0, stream>>>(outCons, nv16);
  sim_kernel<<<dim3(NNN / 128, NNN / 64, BB), 256, 0, stream>>>(nv16, outSim);
  mask_any_kernel<<<1, 256, 0, stream>>>(mask, MMM, hasflag);
  topk_kernel<<<MMM / 8, 256, 0, stream>>>(outSim, mmask, hasflag, allowS, allowU);

  // ---- masked aggregators ----
  auto aggregator = [&](const ushort_t* inw, const float* inb,
                        const ushort_t* ow, const float* ob,
                        const ushort_t* w1, const float* b1, const float* g1, const float* be1,
                        const ushort_t* w2, const float* b2, const float* g2, const float* be2,
                        const ushort_t* w3, const float* b3,
                        const unsigned* allow, int mode,
                        float* addOutF, ushort_t* addOutH) {
    gemm(x16, inw,                 inb,           nullptr, q16, EEE, EEE, 0);
    gemm(x16, inw + EEE * EEE,     inb + EEE,     nullptr, k16, EEE, EEE, 0);
    gemm(x16, inw + 2 * EEE * EEE, inb + 2 * EEE, nullptr, v16, EEE, EEE, 0);
    attn_kernel<<<1024, 128, 0, stream>>>(q16, k16, v16, nullptr, allow, mode, o16);
    gemm(o16, ow, ob, nullptr, ao16, EEE, EEE, 0);
    gemm(ao16, w1, b1, hbuf, nullptr, 2 * EEE, EEE, 0);
    ln_kernel<<<MMM / 8, 256, 0, stream>>>(hbuf, nullptr, g1, be1,
                                           nullptr, h16, 2 * EEE, 1);
    gemm(h16, w2, b2, hbuf2, nullptr, 2 * EEE, 2 * EEE, 0);
    ln_kernel<<<MMM / 8, 256, 0, stream>>>(hbuf2, nullptr, g2, be2,
                                           nullptr, h216, 2 * EEE, 1);
    gemm(h216, w3, b3, fbuf, nullptr, EEE, 2 * EEE, 0);
    add_kernel<<<nElem / 256, 256, 0, stream>>>(xbuf, fbuf, addOutF, addOutH, nElem);
  };

  aggregator(w_a1_in, a1_in_b, w_a1_out, a1_out_b,
             w_m1_1, m1_b1, m1_g1, m1_be1, w_m1_2, m1_b2, m1_g2, m1_be2,
             w_m1_3, m1_b3, allowS, 1, xbuf, x16);
  aggregator(w_a2_in, a2_in_b, w_a2_out, a2_out_b,
             w_m2_1, m2_b1, m2_g1, m2_be1, w_m2_2, m2_b2, m2_g2, m2_be2,
             w_m2_3, m2_b3, allowU, 2, outFeats, nullptr);
}
